// SelfAttention1d_36069135352062
// MI455X (gfx1250) — matrix operand order repaired and checked by reading, not compiled, not run on hardware
//
#include <hip/hip_runtime.h>
#include <hip/hip_bf16.h>

typedef __attribute__((ext_vector_type(16))) _Float16 v16h;
typedef __attribute__((ext_vector_type(8)))  _Float16 v8h;
typedef __attribute__((ext_vector_type(8)))  float    v8f;

#define B_DIM 2
#define C_DIM 1024
#define S_DIM 2048
#define H_DIM 16
#define D_DIM 64
#define M_DIM (B_DIM * S_DIM)   // 4096 tokens
#define NBLK  256               // partial-reduction blocks per batch
#define EPSF  1e-5f
#define LDSP  40                // padded P-row stride (halves): 20*m mod 64 injective

// ---------------------------------------------------------------------------
// Stage 1: per-batch partial sums for mean/var over C*S elements
// ---------------------------------------------------------------------------
__global__ void stats_partial(const float* __restrict__ hs, float* __restrict__ part) {
    const int b = blockIdx.y;
    const float* p = hs + (size_t)b * C_DIM * S_DIM;
    const int N = C_DIM * S_DIM;
    float s = 0.f, ss = 0.f;
    for (int i = blockIdx.x * blockDim.x + threadIdx.x; i < N; i += gridDim.x * blockDim.x) {
        float v = p[i];
        s += v;
        ss += v * v;
    }
    __shared__ float sh[256], sh2[256];
    sh[threadIdx.x] = s;
    sh2[threadIdx.x] = ss;
    __syncthreads();
    for (int o = 128; o > 0; o >>= 1) {
        if (threadIdx.x < o) {
            sh[threadIdx.x]  += sh[threadIdx.x + o];
            sh2[threadIdx.x] += sh2[threadIdx.x + o];
        }
        __syncthreads();
    }
    if (threadIdx.x == 0) {
        part[b * NBLK + blockIdx.x]                 = sh[0];
        part[B_DIM * NBLK + b * NBLK + blockIdx.x]  = sh2[0];
    }
}

__global__ void stats_final(const float* __restrict__ part, float* __restrict__ stats) {
    const int b = blockIdx.x;
    __shared__ float sh[256], sh2[256];
    sh[threadIdx.x]  = part[b * NBLK + threadIdx.x];
    sh2[threadIdx.x] = part[B_DIM * NBLK + b * NBLK + threadIdx.x];
    __syncthreads();
    for (int o = 128; o > 0; o >>= 1) {
        if (threadIdx.x < o) {
            sh[threadIdx.x]  += sh[threadIdx.x + o];
            sh2[threadIdx.x] += sh2[threadIdx.x + o];
        }
        __syncthreads();
    }
    if (threadIdx.x == 0) {
        const float N = (float)(C_DIM * S_DIM);
        float mean = sh[0] / N;
        float var  = sh2[0] / N - mean * mean;
        stats[b * 2 + 0] = mean;
        stats[b * 2 + 1] = rsqrtf(var + EPSF);
    }
}

// ---------------------------------------------------------------------------
// Normalize + affine + transpose (B,C,S) -> token-major (B*S, C) in f16
// ---------------------------------------------------------------------------
__global__ void normalize_f16(const float* __restrict__ hs, const float* __restrict__ gw,
                              const float* __restrict__ gb, const float* __restrict__ stats,
                              _Float16* __restrict__ xn) {
    int idx = blockIdx.x * blockDim.x + threadIdx.x;       // b*S*C + s*C + c
    if (idx >= B_DIM * S_DIM * C_DIM) return;
    int c = idx & (C_DIM - 1);
    int s = (idx >> 10) & (S_DIM - 1);
    int b = idx >> 21;
    float mean = stats[b * 2 + 0];
    float rstd = stats[b * 2 + 1];
    float v = hs[((size_t)b * C_DIM + c) * S_DIM + s];
    xn[idx] = (_Float16)((v - mean) * rstd * gw[c] + gb[c]);
}

__global__ void to_f16(const float* __restrict__ src, _Float16* __restrict__ dst, int n) {
    int i = blockIdx.x * blockDim.x + threadIdx.x;
    if (i < n) dst[i] = (_Float16)src[i];
}

// ---------------------------------------------------------------------------
// WMMA GEMM: out = A(M x 1024) @ W(1024 x 1024)^T + bias, templated epilogue.
// MODE 0: Q/K  -> f16 (B,H,S,D), value scaled by `scale`
// MODE 1: V    -> f16 transposed (B,H,D,S)
// MODE 2: proj -> f32 (B,C,S) with fused residual add
// 4 waves per block, each wave owns one 16x16 output tile.
// ---------------------------------------------------------------------------
template <int MODE>
__global__ void wmma_gemm(const _Float16* __restrict__ A, const _Float16* __restrict__ W,
                          const float* __restrict__ bias, void* __restrict__ outp,
                          const float* __restrict__ residual, float scale) {
    const int lane  = threadIdx.x & 31;
    const int wave  = threadIdx.x >> 5;
    const int mbase = blockIdx.x * 16;
    const int nbase = (blockIdx.y * 4 + wave) * 16;
    const int m  = lane & 15;
    const int hl = lane >> 4;

    const _Float16* arow = A + (size_t)(mbase + m) * C_DIM;
    const _Float16* wrow = W + (size_t)(nbase + m) * C_DIM;

    v8f acc = {};
    for (int kb = 0; kb < C_DIM; kb += 32) {
        // A fragment (16x32 f16): lane m holds row m; halves select K groups
        v8h lo = *(const v8h*)(arow + kb + 16 * hl);
        v8h hi = *(const v8h*)(arow + kb + 16 * hl + 8);
        v16h a;
#pragma unroll
        for (int i = 0; i < 8; ++i) { a[i] = lo[i]; a[i + 8] = hi[i]; }
        // B fragment (32x16): B[k][n] = W[n][k]; lane n reads contiguous K
        v16h bfrag = *(const v16h*)(wrow + kb + 16 * hl);
        acc = __builtin_amdgcn_wmma_f32_16x16x32_f16(false, a, false, bfrag,
                                                     (short)0, acc, false, false);
    }

    const int ncol = nbase + m;
    const float bb = bias[ncol];
#pragma unroll
    for (int r = 0; r < 8; ++r) {
        int tok  = mbase + r + 8 * hl;       // C layout: row r+8*hl, col = lane&15
        int bidx = tok >> 11;                // token / S
        int s    = tok & (S_DIM - 1);
        float v  = acc[r] + bb;
        if (MODE == 0) {
            int hh = ncol >> 6, d = ncol & 63;
            ((_Float16*)outp)[(((size_t)bidx * H_DIM + hh) * S_DIM + s) * D_DIM + d] =
                (_Float16)(v * scale);
        } else if (MODE == 1) {
            int hh = ncol >> 6, d = ncol & 63;
            ((_Float16*)outp)[(((size_t)bidx * H_DIM + hh) * D_DIM + d) * S_DIM + s] =
                (_Float16)v;
        } else {
            size_t oi = ((size_t)bidx * C_DIM + ncol) * S_DIM + s;
            ((float*)outp)[oi] = v + residual[oi];
        }
    }
}

// ---------------------------------------------------------------------------
// Flash-attention style kernel. One wave = 16 query rows of one (b,h).
// Streams keys 32 at a time: 4 WMMAs for scores (K=D=64), online softmax,
// P staged via LDS into A-fragment layout, 4 WMMAs for P@V (D=64).
// launch_bounds(128,1): allow full VGPR budget, no spills in the hot loop.
// ---------------------------------------------------------------------------
__global__ __launch_bounds__(128, 1)
void attention_kernel(const _Float16* __restrict__ Q,  // (B,H,S,D)
                      const _Float16* __restrict__ K,  // (B,H,S,D)
                      const _Float16* __restrict__ Vt, // (B,H,D,S)
                      const int* __restrict__ mask,    // (B,S)
                      _Float16* __restrict__ O) {      // (B*S, C)
    __shared__ _Float16 lds_p[4][16][LDSP];
    const int lane = threadIdx.x & 31;
    const int wave = threadIdx.x >> 5;
    const int tile = blockIdx.x * 4 + wave;       // global query tile
    const int QT   = S_DIM / 16;
    const int bh   = tile / QT;
    const int qt   = tile % QT;
    const int b    = bh / H_DIM;
    const int hh   = bh % H_DIM;
    const int m    = lane & 15;
    const int hl   = lane >> 4;

    const _Float16* qbase = Q  + ((size_t)bh * S_DIM + qt * 16) * D_DIM;
    const _Float16* kbase = K  + (size_t)bh * S_DIM * D_DIM;
    const _Float16* vbase = Vt + (size_t)bh * D_DIM * S_DIM;
    const int* mrow = mask + b * S_DIM;

    // Q A-fragments for K-chunks d=[0,32) and d=[32,64)
    v16h aq[2];
#pragma unroll
    for (int t = 0; t < 2; ++t) {
        const _Float16* qr = qbase + m * D_DIM + t * 32;
        v8h lo = *(const v8h*)(qr + 16 * hl);
        v8h hi = *(const v8h*)(qr + 16 * hl + 8);
#pragma unroll
        for (int i = 0; i < 8; ++i) { aq[t][i] = lo[i]; aq[t][i + 8] = hi[i]; }
    }

    float mrun[8], lrun[8];
    v8f oacc[4] = {};
#pragma unroll
    for (int r = 0; r < 8; ++r) { mrun[r] = -1e30f; lrun[r] = 0.f; }

    for (int kt = 0; kt < S_DIM; kt += 32) {
        v8f c0 = {}, c1 = {};
#pragma unroll
        for (int kb = 0; kb < 2; ++kb) {
            v16h bk0 = *(const v16h*)(kbase + (size_t)(kt + m) * D_DIM + kb * 32 + 16 * hl);
            v16h bk1 = *(const v16h*)(kbase + (size_t)(kt + 16 + m) * D_DIM + kb * 32 + 16 * hl);
            c0 = __builtin_amdgcn_wmma_f32_16x16x32_f16(false, aq[kb], false, bk0,
                                                        (short)0, c0, false, false);
            c1 = __builtin_amdgcn_wmma_f32_16x16x32_f16(false, aq[kb], false, bk1,
                                                        (short)0, c1, false, false);
        }

        // mask (lane holds one key column per tile)
        const int mk0 = mrow[kt + m];
        const int mk1 = mrow[kt + 16 + m];
#pragma unroll
        for (int r = 0; r < 8; ++r) {
            c0[r] = mk0 ? c0[r] : -1e30f;
            c1[r] = mk1 ? c1[r] : -1e30f;
        }

        // row max over 32 keys: butterfly across the 16 lanes of each half
        float mx[8];
#pragma unroll
        for (int r = 0; r < 8; ++r) mx[r] = fmaxf(c0[r], c1[r]);
        for (int off = 1; off < 16; off <<= 1) {
#pragma unroll
            for (int r = 0; r < 8; ++r) mx[r] = fmaxf(mx[r], __shfl_xor(mx[r], off, 32));
        }

        // online-softmax update; mx is reused as the per-row partial sum
#pragma unroll
        for (int r = 0; r < 8; ++r) {
            float mn = fmaxf(mrun[r], mx[r]);
            float al = __expf(mrun[r] - mn);
            mrun[r]  = mn;
            c0[r] = __expf(c0[r] - mn);
            c1[r] = __expf(c1[r] - mn);
            mx[r] = c0[r] + c1[r];
            lrun[r] *= al;
#pragma unroll
            for (int t = 0; t < 4; ++t) oacc[t][r] *= al;
        }
        for (int off = 1; off < 16; off <<= 1) {
#pragma unroll
            for (int r = 0; r < 8; ++r) mx[r] += __shfl_xor(mx[r], off, 32);
        }
#pragma unroll
        for (int r = 0; r < 8; ++r) lrun[r] += mx[r];

        // C-layout -> A-fragment reshape of P through LDS.
        // DS ops of a wave complete in order, so the cross-lane write->read
        // needs no explicit barrier; compiler inserts the dscnt wait for use.
#pragma unroll
        for (int r = 0; r < 8; ++r) {
            lds_p[wave][r + 8 * hl][m]      = (_Float16)c0[r];
            lds_p[wave][r + 8 * hl][16 + m] = (_Float16)c1[r];
        }
        // A-fragment read: two contiguous 8-half runs per lane (b128 each)
        v8h plo = *(const v8h*)&lds_p[wave][m][16 * hl];
        v8h phi = *(const v8h*)&lds_p[wave][m][16 * hl + 8];
        v16h ap;
#pragma unroll
        for (int i = 0; i < 8; ++i) { ap[i] = plo[i]; ap[i + 8] = phi[i]; }

        // O += P @ V  (K=32 keys, four 16-wide d tiles); Vt contiguous along keys
#pragma unroll
        for (int t = 0; t < 4; ++t) {
            v16h bv = *(const v16h*)(vbase + (size_t)(t * 16 + m) * S_DIM + kt + 16 * hl);
            oacc[t] = __builtin_amdgcn_wmma_f32_16x16x32_f16(false, ap, false, bv,
                                                             (short)0, oacc[t], false, false);
        }
    }

    // epilogue: divide by row sums, store token-major f16 (B*S, C)
#pragma unroll
    for (int r = 0; r < 8; ++r) {
        int row = qt * 16 + r + 8 * hl;
        float inv = 1.f / lrun[r];
#pragma unroll
        for (int t = 0; t < 4; ++t) {
            O[((size_t)b * S_DIM + row) * C_DIM + hh * D_DIM + t * 16 + m] =
                (_Float16)(oacc[t][r] * inv);
        }
    }
}

// ---------------------------------------------------------------------------
extern "C" void kernel_launch(void* const* d_in, const int* in_sizes, int n_in,
                              void* d_out, int out_size, void* d_ws, size_t ws_size,
                              hipStream_t stream) {
    (void)in_sizes; (void)n_in; (void)out_size; (void)ws_size;
    const float* hs = (const float*)d_in[0];
    const int*   am = (const int*)  d_in[1];
    const float* gw = (const float*)d_in[2];
    const float* gb = (const float*)d_in[3];
    const float* Wq = (const float*)d_in[4];
    const float* bq = (const float*)d_in[5];
    const float* Wk = (const float*)d_in[6];
    const float* bk = (const float*)d_in[7];
    const float* Wv = (const float*)d_in[8];
    const float* bv = (const float*)d_in[9];
    const float* Wo = (const float*)d_in[10];
    const float* bo = (const float*)d_in[11];

    char* ws = (char*)d_ws;
    float* part  = (float*)(ws);
    float* stats = (float*)(ws + (16 << 10));
    _Float16* xn = (_Float16*)(ws + (32 << 10));
    const size_t TOK = (size_t)M_DIM * C_DIM;      // 4M halves per activation buffer
    _Float16* qb   = xn   + TOK;
    _Float16* kbuf = qb   + TOK;
    _Float16* vbuf = kbuf + TOK;
    _Float16* attn = vbuf + TOK;
    _Float16* wqh  = attn + TOK;
    const size_t WSZ = (size_t)C_DIM * C_DIM;
    _Float16* wkh = wqh + WSZ;
    _Float16* wvh = wkh + WSZ;
    _Float16* woh = wvh + WSZ;

    // 1) stats + normalize
    stats_partial<<<dim3(NBLK, B_DIM), 256, 0, stream>>>(hs, part);
    stats_final<<<B_DIM, 256, 0, stream>>>(part, stats);
    normalize_f16<<<(B_DIM * S_DIM * C_DIM) / 256, 256, 0, stream>>>(hs, gw, gb, stats, xn);

    // 2) weights to f16
    const int wn = C_DIM * C_DIM;
    to_f16<<<wn / 256, 256, 0, stream>>>(Wq, wqh, wn);
    to_f16<<<wn / 256, 256, 0, stream>>>(Wk, wkh, wn);
    to_f16<<<wn / 256, 256, 0, stream>>>(Wv, wvh, wn);
    to_f16<<<wn / 256, 256, 0, stream>>>(Wo, woh, wn);

    // 3) QKV projections (WMMA)
    const float scale = 0.35355339059327373f;   // 1/sqrt(sqrt(D)), D=64
    dim3 ggrid(M_DIM / 16, (C_DIM / 16) / 4);
    wmma_gemm<0><<<ggrid, 128, 0, stream>>>(xn, wqh, bq, qb,   nullptr, scale);
    wmma_gemm<0><<<ggrid, 128, 0, stream>>>(xn, wkh, bk, kbuf, nullptr, scale);
    wmma_gemm<1><<<ggrid, 128, 0, stream>>>(xn, wvh, bv, vbuf, nullptr, 1.0f);

    // 4) attention (WMMA flash)
    attention_kernel<<<(B_DIM * H_DIM * (S_DIM / 16)) / 4, 128, 0, stream>>>(
        qb, kbuf, vbuf, am, attn);

    // 5) output projection + residual, written directly as (B,C,S) f32
    wmma_gemm<2><<<ggrid, 128, 0, stream>>>(attn, woh, bo, d_out, hs, 1.0f);
}
